// CURVGG8_68813966016896
// MI455X (gfx1250) — compile-verified, matrix operand order checked by reading, not run on hardware
//
#include <hip/hip_runtime.h>
#include <cstdint>
#include <cstddef>

// ---------------------------------------------------------------------------
// CUR-VGG8 forward for MI455X (gfx1250, wave32).
//  - Big GEMMs (conv implicit-im2col + FC chain): v_wmma_f32_16x16x32_bf16
//    with LDS double buffering; FC A-tiles staged with
//    global_load_async_to_lds_b128 when available (ASYNCcnt discipline).
//  - pinv via Newton-Schulz in f32:               v_wmma_f32_16x16x4_f32
//  - conv bias dropped (cancelled exactly by train-mode BatchNorm)
// ---------------------------------------------------------------------------

typedef __bf16 bf16;
typedef __attribute__((ext_vector_type(16))) __bf16 v16bf;
typedef __attribute__((ext_vector_type(8)))  __bf16 v8bf;
typedef __attribute__((ext_vector_type(8)))  float  v8f;
typedef __attribute__((ext_vector_type(2)))  float  v2f;
typedef __attribute__((ext_vector_type(4)))  int    v4i;

typedef __attribute__((address_space(1))) v4i* gv4i_p;   // global int4*
typedef __attribute__((address_space(3))) v4i* lv4i_p;   // LDS int4*

#if __has_builtin(__builtin_amdgcn_wmma_f32_16x16x4_f32)
#define HAVE_WMMA_F32X4 1
#else
#define HAVE_WMMA_F32X4 0
#endif

#if __has_builtin(__builtin_amdgcn_global_load_async_to_lds_b128)
#define HAVE_ASYNC_LDS 1
#else
#define HAVE_ASYNC_LDS 0
#endif

#if __has_builtin(__builtin_amdgcn_s_wait_asynccnt)
#define WAIT_ASYNC0() __builtin_amdgcn_s_wait_asynccnt(0)
#else
#define WAIT_ASYNC0() asm volatile("s_wait_asynccnt 0" ::: "memory")
#endif

// Build a 16-element bf16 fragment from an LDS row: K {k0..k0+7, k0+16..k0+23}
// (the wave32 16-bit A/B operand layout; k0 = 0 for lanes 0-15, 8 for 16-31).
__device__ __forceinline__ v16bf frag16(const bf16* rowp, int k0)
{
  v8bf lo = *(const v8bf*)(rowp + k0);
  v8bf hi = *(const v8bf*)(rowp + k0 + 16);
  return __builtin_shufflevector(lo, hi, 0, 1, 2, 3, 4, 5, 6, 7,
                                 8, 9, 10, 11, 12, 13, 14, 15);
}

// ---------------------------------------------------------------------------
// Small f32 GEMM with WMMA 16x16x4:  D = alpha*(A@B) + beta*C
// A[M,K], B[K,N], C/D[M,N] row-major. One wave per block -> one 16x16 tile.
// N,K multiples of 16/4; M may be ragged (clamp + mask, no EXEC divergence
// in the k-loop so WMMA always sees EXEC == all-ones).
// ---------------------------------------------------------------------------
__global__ void gemm_f32_wmma(const float* __restrict__ A, const float* __restrict__ B,
                              const float* __restrict__ Cin, float* __restrict__ D,
                              int M, int N, int K, float alpha, float beta)
{
  const int lane = threadIdx.x;
  const int half = lane >> 4, l = lane & 15;
  const int r0 = blockIdx.y * 16, c0 = blockIdx.x * 16;
  const int ra = r0 + l;
  const int rc = ra < M ? ra : (M - 1);          // clamped row (always in range)
  const float am = ra < M ? 1.f : 0.f;           // mask instead of branch
  const float* __restrict__ Arow = A + (size_t)rc * K + 2 * half;
  const float* __restrict__ Bc   = B + c0 + l;
  v8f acc = {};
#if HAVE_WMMA_F32X4
  for (int kb = 0; kb < K; kb += 4) {
    v2f a, b;
    a[0] = Arow[kb] * am;
    a[1] = Arow[kb + 1] * am;
    b[0] = Bc[(size_t)(kb + 2 * half) * N];
    b[1] = Bc[(size_t)(kb + 2 * half + 1) * N];
    acc = __builtin_amdgcn_wmma_f32_16x16x4_f32(false, a, false, b, (short)0, acc,
                                                false, false);
  }
#else
  #pragma unroll
  for (int i = 0; i < 8; ++i) {
    int r = r0 + half * 8 + i;
    float s = 0.f;
    if (r < M) for (int k = 0; k < K; ++k) s += A[(size_t)r * K + k] * B[(size_t)k * N + c0 + l];
    acc[i] = s;
  }
#endif
  #pragma unroll
  for (int i = 0; i < 8; ++i) {
    const int r = r0 + half * 8 + i;   // C/D layout: VGPR i -> row i (lanes<16) / i+8
    if (r < M) {
      float v = alpha * acc[i];
      if (Cin) v += beta * Cin[(size_t)r * N + c0 + l];
      D[(size_t)r * N + c0 + l] = v;
    }
  }
}

// ---------------------------------------------------------------------------
// bf16 GEMM with WMMA 16x16x32 (f32 accumulate), optional on-the-fly im2col
// for 3x3 SAME conv. A[M,K] bf16 (or NCHW activation in im2col mode),
// B[K,N] bf16 row-major, D[M,N] f32 or bf16.
// 256 threads = 8 waves; block tile 128x64, BK=32, LDS double buffered (24KB).
// flags: bit0 = relu, bit1 = write bf16.
// ---------------------------------------------------------------------------
template <bool IM2COL>
__global__ void gemm_bf16_wmma(const bf16* __restrict__ A, const bf16* __restrict__ Bm,
                               const float* __restrict__ bias, void* __restrict__ Dst,
                               int M, int N, int K, int H, int W, int C, int flags)
{
  __shared__ __align__(16) bf16 As[2][128][32];  // [buf][m][k]  16 KB
  __shared__ __align__(16) bf16 Bs[2][64][32];   // [buf][n][k]   8 KB (transposed)
  const int tid  = threadIdx.x;
  const int bm   = blockIdx.y * 128, bn = blockIdx.x * 64;
  const int wave = tid >> 5, lane = tid & 31;
  const int wm   = wave >> 1, wn = wave & 1;
  const int half = lane >> 4, l = lane & 15;

  // --- staging assignments ---
  const int arow = tid >> 1, akoff = (tid & 1) * 16;   // A: 16 contig k per thread
  const int gmA  = bm + arow;
  int b_ = 0, oh = 0, ow = 0;
  if (IM2COL) {
    const int hw = H * W;
    b_ = gmA / hw;
    const int rem = gmA - b_ * hw;
    oh = rem / W;
    ow = rem - oh * W;
  }
  const int bk = tid >> 3, bn0 = (tid & 7) * 8;        // B: 8 contig n per thread

  auto stageA = [&](int kb, int p) {
    if (IM2COL) {
      #pragma unroll
      for (int j = 0; j < 16; ++j) {
        const int k = kb + akoff + j;
        bf16 v = (bf16)0.0f;
        if (k < K) {
          const int c  = k / 9, r9 = k - c * 9;
          const int kh = r9 / 3, kw = r9 - kh * 3;
          const int ih = oh + kh - 1, iw = ow + kw - 1;
          if ((unsigned)ih < (unsigned)H && (unsigned)iw < (unsigned)W)
            v = A[((size_t)(b_ * C + c) * H + ih) * W + iw];
        }
        As[p][arow][akoff + j] = v;
      }
    } else {
#if HAVE_ASYNC_LDS
      // per-lane 16B global->LDS async copies (FC tiles are dense: K%32==0)
      #pragma unroll
      for (int cc = 0; cc < 2; ++cc) {
        const int ch = tid + cc * 256;                 // 512 x 16B chunks = 8 KB
        const int r = ch >> 2, ko = (ch & 3) * 8;
        void* srcv = (void*)(A + (size_t)(bm + r) * K + kb + ko);  // drop const
        __builtin_amdgcn_global_load_async_to_lds_b128(
            (gv4i_p)srcv, (lv4i_p)&As[p][r][ko], 0, 0);
      }
#else
      const v8bf* gp = (const v8bf*)(A + (size_t)gmA * K + kb + akoff);
      *(v8bf*)&As[p][arow][akoff]     = gp[0];
      *(v8bf*)&As[p][arow][akoff + 8] = gp[1];
#endif
    }
  };

  auto stageB = [&](int kb, int p) {
    v8bf bv = {};
    if (kb + bk < K) bv = *(const v8bf*)(Bm + (size_t)(kb + bk) * N + bn + bn0);
    #pragma unroll
    for (int j = 0; j < 8; ++j) Bs[p][bn0 + j][bk] = bv[j];   // transpose into LDS
    if (kb + 32 + bk < K)   // pull K-slice after next toward L2 (global_prefetch_b8)
      __builtin_prefetch((const void*)(Bm + (size_t)(kb + 32 + bk) * N + bn + bn0), 0, 1);
  };

  v8f acc[2][2] = {};
  stageA(0, 0);
  stageB(0, 0);

  for (int kb = 0; kb < K; kb += 32) {
    const int p = (kb >> 5) & 1;
#if HAVE_ASYNC_LDS
    if (!IM2COL) WAIT_ASYNC0();      // this wave's async copies into buf p done
#endif
    __syncthreads();                 // all waves' staging of buf p published
    if (kb + 32 < K) { stageA(kb + 32, p ^ 1); stageB(kb + 32, p ^ 1); }

    const int k0 = half ? 8 : 0;     // wave32 16-bit operand K split
    #pragma unroll
    for (int mi = 0; mi < 2; ++mi) {
      const v16bf afr = frag16(&As[p][wm * 32 + mi * 16 + l][0], k0);
      #pragma unroll
      for (int ni = 0; ni < 2; ++ni) {
        const v16bf bfr = frag16(&Bs[p][wn * 32 + ni * 16 + l][0], k0);
        acc[mi][ni] = __builtin_amdgcn_wmma_f32_16x16x32_bf16(
            false, afr, false, bfr, (short)0, acc[mi][ni], false, false);
      }
    }
  }

  // ---- epilogue: optional bias / relu / bf16 store ----
  const bool relu = flags & 1, obf = flags & 2;
  #pragma unroll
  for (int mi = 0; mi < 2; ++mi)
    #pragma unroll
    for (int ni = 0; ni < 2; ++ni)
      #pragma unroll
      for (int i = 0; i < 8; ++i) {
        const int r  = bm + wm * 32 + mi * 16 + half * 8 + i;
        const int cI = bn + wn * 32 + ni * 16 + l;
        float v = acc[mi][ni][i];
        if (bias) v += bias[cI];
        if (relu) v = v > 0.f ? v : 0.f;
        if (r < M) {
          if (obf) ((bf16*)Dst)[(size_t)r * N + cI] = (bf16)v;
          else     ((float*)Dst)[(size_t)r * N + cI] = v;
        }
      }
}

// ---------------------------------------------------------------------------
// Helper kernels
// ---------------------------------------------------------------------------
__global__ void gather_avg(const float* __restrict__ wc, const float* __restrict__ wr,
                           const int* __restrict__ rows, const int* __restrict__ cols,
                           float* __restrict__ Am, int n, int wc_ld, int wr_ld)
{
  const int idx = blockIdx.x * 256 + threadIdx.x;
  if (idx < n * n) {
    const int i = idx / n, j = idx - i * n;
    Am[idx] = 0.5f * (wc[(size_t)rows[i] * wc_ld + j] + wr[(size_t)i * wr_ld + cols[j]]);
  }
}

// alpha = 1 / (||A||_1 * ||A||_inf), single block
__global__ void norm_alpha(const float* __restrict__ Am, float* __restrict__ alpha, int n)
{
  __shared__ float red[256];
  const int tid = threadIdx.x;
  float cmax = 0.f, rmax = 0.f;
  for (int j = tid; j < n; j += 256) {
    float s = 0.f;
    for (int i = 0; i < n; ++i) s += fabsf(Am[(size_t)i * n + j]);
    cmax = fmaxf(cmax, s);
  }
  for (int i = tid; i < n; i += 256) {
    float s = 0.f;
    for (int j = 0; j < n; ++j) s += fabsf(Am[(size_t)i * n + j]);
    rmax = fmaxf(rmax, s);
  }
  red[tid] = cmax; __syncthreads();
  for (int o = 128; o; o >>= 1) { if (tid < o) red[tid] = fmaxf(red[tid], red[tid + o]); __syncthreads(); }
  const float n1 = red[0]; __syncthreads();
  red[tid] = rmax; __syncthreads();
  for (int o = 128; o; o >>= 1) { if (tid < o) red[tid] = fmaxf(red[tid], red[tid + o]); __syncthreads(); }
  if (tid == 0) alpha[0] = 1.f / (n1 * red[0]);
}

__global__ void transpose_scale(const float* __restrict__ Am, const float* __restrict__ alpha,
                                float* __restrict__ X, int n)
{
  const int idx = blockIdx.x * 256 + threadIdx.x;
  if (idx < n * n) {
    const int i = idx / n, j = idx - i * n;
    X[idx] = alpha[0] * Am[(size_t)j * n + i];   // X0 = a * A^T
  }
}

__global__ void cvt_bf16_k(const float* __restrict__ src, bf16* __restrict__ dst, size_t n)
{
  const size_t i = (size_t)blockIdx.x * 256 + threadIdx.x;
  if (i < n) dst[i] = (bf16)src[i];
}

// per-channel sum / sumsq over the M rows of y[M,F] (bf16)
__global__ void stats_kernel(const bf16* __restrict__ y, float* __restrict__ sums,
                             float* __restrict__ sqs, int Mrows, int F)
{
  const int f = blockIdx.x, tid = threadIdx.x;
  float s = 0.f, s2 = 0.f;
  for (int m = tid; m < Mrows; m += blockDim.x) {
    const float v = (float)y[(size_t)m * F + f];
    s += v; s2 += v * v;
  }
  __shared__ float sh[256], sh2[256];
  sh[tid] = s; sh2[tid] = s2; __syncthreads();
  for (int o = 128; o; o >>= 1) {
    if (tid < o) { sh[tid] += sh[tid + o]; sh2[tid] += sh2[tid + o]; }
    __syncthreads();
  }
  if (tid == 0) { sums[f] = sh[0]; sqs[f] = sh2[0]; }
}

// BN(train, biased var) + ReLU + optional 2x2 maxpool; y[M=B*H*W, F] -> NCHW bf16
__global__ void bn_relu_pool(const bf16* __restrict__ y, const float* __restrict__ sums,
                             const float* __restrict__ sqs, const float* __restrict__ gamma,
                             const float* __restrict__ beta, bf16* __restrict__ out,
                             int Bn, int F, int H, int W, int pool, float invM)
{
  const int Ho = pool ? H / 2 : H, Wo = pool ? W / 2 : W;
  const size_t total = (size_t)Bn * F * Ho * Wo;
  const size_t idx = (size_t)blockIdx.x * 256 + threadIdx.x;
  if (idx >= total) return;
  const int wo = (int)(idx % Wo);
  size_t t = idx / Wo;
  const int ho = (int)(t % Ho); t /= Ho;
  const int f  = (int)(t % F);
  const int b  = (int)(t / F);
  const float mean = sums[f] * invM;
  const float var  = sqs[f] * invM - mean * mean;
  const float sc   = gamma[f] * rsqrtf(var + 1e-5f);
  const float sh   = beta[f] - mean * sc;
  float r;
  if (pool) {
    const int ih = ho * 2, iw = wo * 2;
    const float a00 = fmaxf((float)y[((size_t)(b * H + ih) * W + iw) * F + f] * sc + sh, 0.f);
    const float a01 = fmaxf((float)y[((size_t)(b * H + ih) * W + iw + 1) * F + f] * sc + sh, 0.f);
    const float a10 = fmaxf((float)y[((size_t)(b * H + ih + 1) * W + iw) * F + f] * sc + sh, 0.f);
    const float a11 = fmaxf((float)y[((size_t)(b * H + ih + 1) * W + iw + 1) * F + f] * sc + sh, 0.f);
    r = fmaxf(fmaxf(a00, a01), fmaxf(a10, a11));
  } else {
    r = fmaxf((float)y[((size_t)(b * H + ho) * W + wo) * F + f] * sc + sh, 0.f);
  }
  out[idx] = (bf16)r;   // idx is already NCHW-linear
}

// final [512,4096] @ [10,4096]^T + b  (N=10, too small for WMMA tiles)
__global__ void fc3_kernel(const bf16* __restrict__ h, const float* __restrict__ w,
                           const float* __restrict__ b, float* __restrict__ out,
                           int Bn, int K, int NC)
{
  const int idx = blockIdx.x * 256 + threadIdx.x;
  if (idx >= Bn * NC) return;
  const int n = idx % NC, m = idx / NC;
  float s = b[n];
  for (int k = 0; k < K; ++k) s += (float)h[(size_t)m * K + k] * w[(size_t)n * K + k];
  out[idx] = s;
}

// ---------------------------------------------------------------------------
// Host orchestration
// ---------------------------------------------------------------------------
extern "C" void kernel_launch(void* const* d_in, const int* in_sizes, int n_in,
                              void* d_out, int out_size, void* d_ws, size_t ws_size,
                              hipStream_t stream)
{
  (void)in_sizes; (void)n_in; (void)out_size; (void)ws_size;

  // d_in order (tuples flattened): x | conv_wc[8] | conv_wr[8] | conv_b[8] |
  // conv_rows[8] | conv_cols[8] | bn_gamma[8] | bn_beta[8] | fc_wc[2] |
  // fc_wr[2] | fc_b[2] | fc_rows[2] | fc_cols[2] | fc3_w | fc3_b
  auto F32 = [&](int i) { return (const float*)d_in[i]; };
  auto I32 = [&](int i) { return (const int*)d_in[i]; };
  const float* x = F32(0);
  auto cwc  = [&](int i) { return F32(1 + i);  };
  auto cwr  = [&](int i) { return F32(9 + i);  };
  auto crow = [&](int i) { return I32(25 + i); };
  auto ccol = [&](int i) { return I32(33 + i); };
  auto gam  = [&](int i) { return F32(41 + i); };
  auto bet  = [&](int i) { return F32(49 + i); };
  auto fwc  = [&](int j) { return F32(57 + j); };
  auto fwr  = [&](int j) { return F32(59 + j); };
  auto fb   = [&](int j) { return F32(61 + j); };
  auto frow = [&](int j) { return I32(63 + j); };
  auto fcol = [&](int j) { return I32(65 + j); };
  const float* fc3w = F32(67);
  const float* fc3b = F32(68);

  char* ws = (char*)d_ws;
  auto at = [&](size_t off) { return (void*)(ws + off); };
  const size_t MB = 1u << 20;
  const size_t OFF_Y     = 0;          // 64 MiB: conv output bf16 [M,F]
  const size_t OFF_ACT_A = 64 * MB;    // 16 MiB activation ping
  const size_t OFF_ACT_B = 80 * MB;    // 16 MiB activation pong
  const size_t OFF_MAT_A = 96 * MB;    // 1 MiB  A (<=512x512 f32)
  const size_t OFF_X0    = 97 * MB;    // 1 MiB  Newton X ping
  const size_t OFF_X1    = 98 * MB;    // 1 MiB  Newton X pong
  const size_t OFF_T     = 99 * MB;    // 1 MiB  Newton T = A@X
  const size_t OFF_TMP   = 100 * MB;   // 1 MiB  U@wr
  const size_t OFF_WEFF  = 101 * MB;   // 10 MiB w_eff f32 (max 4608x512)
  const size_t OFF_WEFFB = 111 * MB;   // 5 MiB  w_eff bf16
  const size_t OFF_SUM   = 116 * MB;   // stats
  const size_t OFF_SQ    = OFF_SUM + 4096;
  const size_t OFF_ALPHA = OFF_SUM + 8192;
  const size_t OFF_UBF   = 117 * MB;   // 2x 512 KiB U bf16
  const size_t OFF_WCBF  = 118 * MB;   // 5 MiB  fc_wc bf16
  const size_t OFF_WRBF  = 123 * MB;   // 8 MiB  fc_wr bf16

  auto g32 = [&](const float* A, const float* B, const float* Cin, float* D,
                 int M, int N, int K, float alpha, float beta) {
    dim3 g(N / 16, (M + 15) / 16);
    gemm_f32_wmma<<<g, 32, 0, stream>>>(A, B, Cin, D, M, N, K, alpha, beta);
  };

  // Newton-Schulz inverse of the n x n intersection block -> returns f32 [n,n]
  auto compute_pinv = [&](const float* wcP, const float* wrP, const int* rowsP,
                          const int* colsP, int n, int wc_ld, int wr_ld) -> float* {
    float* Am = (float*)at(OFF_MAT_A);
    float* Xa = (float*)at(OFF_X0);
    float* Xb = (float*)at(OFF_X1);
    float* Tm = (float*)at(OFF_T);
    float* al = (float*)at(OFF_ALPHA);
    const int nn = n * n;
    gather_avg<<<(nn + 255) / 256, 256, 0, stream>>>(wcP, wrP, rowsP, colsP, Am, n, wc_ld, wr_ld);
    norm_alpha<<<1, 256, 0, stream>>>(Am, al, n);
    transpose_scale<<<(nn + 255) / 256, 256, 0, stream>>>(Am, al, Xa, n);
    for (int it = 0; it < 40; ++it) {
      g32(Am, Xa, nullptr, Tm, n, n, n, 1.f, 0.f);       // T  = A @ X
      g32(Xa, Tm, Xa, Xb, n, n, n, -1.f, 2.f);           // X' = 2X - X @ T
      float* t = Xa; Xa = Xb; Xb = t;
    }
    return Xa;
  };

  bf16* actA = (bf16*)at(OFF_ACT_A);
  bf16* actB = (bf16*)at(OFF_ACT_B);
  bf16* ybuf = (bf16*)at(OFF_Y);
  float* sums = (float*)at(OFF_SUM);
  float* sqs  = (float*)at(OFF_SQ);

  // x -> bf16 NCHW
  const size_t xcnt = (size_t)512 * 3 * 32 * 32;
  cvt_bf16_k<<<(unsigned)((xcnt + 255) / 256), 256, 0, stream>>>(x, actA, xcnt);
  bf16* cur = actA;
  bf16* nxt = actB;

  struct CC { int C, F, H, n, pool; };
  const CC cc[8] = {{3, 64, 32, 16, 1},   {64, 128, 16, 64, 1},
                    {128, 256, 8, 128, 0},{256, 256, 8, 128, 1},
                    {256, 512, 4, 128, 0},{512, 512, 4, 128, 1},
                    {512, 512, 2, 256, 0},{512, 512, 2, 256, 1}};

  for (int i = 0; i < 8; ++i) {
    const CC& c = cc[i];
    const int kwr = c.C * 9;
    // --- effective weight: w_eff = wc @ inv(A) @ wr  (f32 WMMA) ---
    float* U    = compute_pinv(cwc(i), cwr(i), crow(i), ccol(i), c.n, c.n, c.F);
    float* tmp  = (float*)at(OFF_TMP);
    float* weff = (float*)at(OFF_WEFF);
    bf16* weffb = (bf16*)at(OFF_WEFFB);
    g32(U, cwr(i), nullptr, tmp, c.n, c.F, c.n, 1.f, 0.f);       // [n,F]
    g32(cwc(i), tmp, nullptr, weff, kwr, c.F, c.n, 1.f, 0.f);    // [C*9,F]
    const size_t wn = (size_t)kwr * c.F;
    cvt_bf16_k<<<(unsigned)((wn + 255) / 256), 256, 0, stream>>>(weff, weffb, wn);
    // --- conv as implicit-im2col bf16 GEMM: y[M,F] ---
    const int M = 512 * c.H * c.H;
    dim3 g(c.F / 64, M / 128);
    gemm_bf16_wmma<true><<<g, 256, 0, stream>>>(cur, weffb, nullptr, (void*)ybuf,
                                                M, c.F, kwr, c.H, c.H, c.C, /*bf16 out*/2);
    // --- BN(train) + ReLU (+ maxpool) ---
    stats_kernel<<<c.F, 256, 0, stream>>>(ybuf, sums, sqs, M, c.F);
    const int Ho = c.pool ? c.H / 2 : c.H;
    const size_t total = (size_t)512 * c.F * Ho * Ho;
    bn_relu_pool<<<(unsigned)((total + 255) / 256), 256, 0, stream>>>(
        ybuf, sums, sqs, gam(i), bet(i), nxt, 512, c.F, c.H, c.H, c.pool, 1.0f / (float)M);
    bf16* t = cur; cur = nxt; nxt = t;
  }
  // cur now holds flattened [512, 512] bf16

  // --- FC weight prep: U_j (bf16) + bf16 copies of wc/wr (w_eff stays factored) ---
  const int fins[2] = {512, 4096};
  bf16* Ubf[2]  = {(bf16*)at(OFF_UBF),  (bf16*)at(OFF_UBF + (size_t)512 * 512 * 2)};
  bf16* wcbf[2] = {(bf16*)at(OFF_WCBF), (bf16*)at(OFF_WCBF + (size_t)512 * 512 * 2)};
  bf16* wrbf[2] = {(bf16*)at(OFF_WRBF), (bf16*)at(OFF_WRBF + (size_t)512 * 4096 * 2)};
  for (int j = 0; j < 2; ++j) {
    float* U = compute_pinv(fwc(j), fwr(j), frow(j), fcol(j), 512, 512, 4096);
    cvt_bf16_k<<<(512 * 512 + 255) / 256, 256, 0, stream>>>(U, Ubf[j], (size_t)512 * 512);
    const size_t wcn = (size_t)fins[j] * 512;
    cvt_bf16_k<<<(unsigned)((wcn + 255) / 256), 256, 0, stream>>>(fwc(j), wcbf[j], wcn);
    const size_t wrn = (size_t)512 * 4096;
    cvt_bf16_k<<<(unsigned)((wrn + 255) / 256), 256, 0, stream>>>(fwr(j), wrbf[j], wrn);
  }

  auto fcgemm = [&](const bf16* A, const bf16* B, const float* bias, bf16* D,
                    int M, int N, int K, int flags) {
    dim3 g(N / 64, M / 128);
    gemm_bf16_wmma<false><<<g, 256, 0, stream>>>(A, B, bias, (void*)D, M, N, K, 0, 0, 0, flags);
  };

  // FC1: relu((x @ wc0) @ U0 @ wr0 + b0)
  fcgemm(cur, wcbf[0], nullptr, nxt, 512, 512, 512, 2);
  fcgemm(nxt, Ubf[0],  nullptr, cur, 512, 512, 512, 2);
  fcgemm(cur, wrbf[0], fb(0),   nxt, 512, 4096, 512, 2 | 1);
  // FC2: relu((h1 @ wc1) @ U1 @ wr1 + b1)
  fcgemm(nxt, wcbf[1], nullptr, cur, 512, 512, 4096, 2);
  fcgemm(cur, Ubf[1],  nullptr, nxt, 512, 512, 512, 2);
  fcgemm(nxt, wrbf[1], fb(1),   cur, 512, 4096, 512, 2 | 1);
  // FC3 -> d_out [512,10] f32
  fc3_kernel<<<(512 * 10 + 255) / 256, 256, 0, stream>>>(cur, fc3w, fc3b, (float*)d_out,
                                                         512, 4096, 10);
}
